// EHRNN_80444737454130
// MI455X (gfx1250) — matrix operand order repaired and checked by reading, not compiled, not run on hardware
//
#include <hip/hip_runtime.h>
#include <math.h>

// ---------------------------------------------------------------------------
// Problem constants (from the reference)
// ---------------------------------------------------------------------------
#define B_ROWS   8192
#define INPUT    2048
#define HIDDEN   2048
#define KPOOL    45
#define NUM_EH   45
#define NJ       3                      // window lengths {32,128,512}
#define EST_DIM  (NUM_EH * NJ * 2)      // 270
#define CLASSES  10
#define LIN_DIM  (HIDDEN + EST_DIM)     // 2318

// GEMM tiling
#define BM 128
#define BN 128
#define BK 32

typedef __attribute__((ext_vector_type(16))) __bf16 v16bf;
typedef __attribute__((ext_vector_type(8)))  float  v8f;

// ---------------------------------------------------------------------------
// TDM availability (device pass only; host pass parses the fallback branch)
// ---------------------------------------------------------------------------
#if defined(__AMDGCN__) && __has_builtin(__builtin_amdgcn_tensor_load_to_lds)
#define USE_TDM 1
#else
#define USE_TDM 0
#endif

#if USE_TDM
typedef unsigned int u32x4 __attribute__((ext_vector_type(4)));
typedef int          i32x4 __attribute__((ext_vector_type(4)));
typedef int          i32x8 __attribute__((ext_vector_type(8)));

// Issue a 2D TDM load: tile of (rows x 32) f32 elements, row stride
// `row_stride_elems`, packed contiguously into LDS at `lds_dst`.
// D# built per CDNA5 ISA 8.3/8.4 (group0: count/lds/global/type,
// group1: data_size, dims, tile dims, dim0 stride).
__device__ __forceinline__ void tdm_load_tile_f32(const float* gsrc, void* lds_dst,
                                                  int rows, int row_stride_elems) {
  unsigned long long ga = (unsigned long long)(uintptr_t)gsrc;
  unsigned lofs = (unsigned)(uintptr_t)lds_dst;   // LDS byte offset = addr[31:0]

  u32x4 g0;
  g0.x = 1u;                                       // count=1, is_restore=0, no gather
  g0.y = lofs;                                     // lds_addr
  g0.z = (unsigned)ga;                             // global_addr[31:0]
  g0.w = (unsigned)((ga >> 32) & 0x01FFFFFFu)      // global_addr[56:32]
       | (2u << 30);                               // type = 2 ("image")

  const unsigned dim0 = (unsigned)row_stride_elems;  // tensor_dim0 (row length)
  const unsigned dim1 = 1u << 20;                    // tensor_dim1 (no OOB clipping)
  i32x8 g1;
  g1[0] = (int)(2u << 16);                                          // data_size=4B
  g1[1] = (int)((dim0 & 0xFFFFu) << 16);                            // dim0[15:0]
  g1[2] = (int)(((dim0 >> 16) & 0xFFFFu) | ((dim1 & 0xFFFFu) << 16)); // dim0 hi, dim1 lo
  g1[3] = (int)(((dim1 >> 16) & 0xFFFFu) | ((unsigned)BK << 16));   // dim1 hi, tile_dim0=32
  g1[4] = rows;                                                     // tile_dim1 (tile_dim2=0)
  g1[5] = (int)dim0;                                                // tensor_dim0_stride lo32
  g1[6] = 0;                                                        // stride hi, dim1_stride lo
  g1[7] = 0;

  i32x4 z4 = {0, 0, 0, 0};
#if __clang_major__ >= 23
  i32x8 z8 = {0, 0, 0, 0, 0, 0, 0, 0};
  __builtin_amdgcn_tensor_load_to_lds(g0, g1, z4, z4, z8, 0);
#else
  __builtin_amdgcn_tensor_load_to_lds(g0, g1, z4, z4, 0);
#endif
}
#endif  // USE_TDM

// ---------------------------------------------------------------------------
// Helpers
// ---------------------------------------------------------------------------
__device__ __forceinline__ void cvt8(v16bf& d, int base, float4 lo, float4 hi) {
  d[base + 0] = (__bf16)lo.x; d[base + 1] = (__bf16)lo.y;
  d[base + 2] = (__bf16)lo.z; d[base + 3] = (__bf16)lo.w;
  d[base + 4] = (__bf16)hi.x; d[base + 5] = (__bf16)hi.y;
  d[base + 6] = (__bf16)hi.z; d[base + 7] = (__bf16)hi.w;
}

__device__ __forceinline__ float wave_reduce_add(float v) {
#pragma unroll
  for (int off = 16; off > 0; off >>= 1) v += __shfl_xor(v, off, 32);
  return v;
}

// ---------------------------------------------------------------------------
// Kernel 1: rnn_out = tanh(X * W_ih^T + b_ih + b_hh)
//   Block = 256 threads (8 waves). Block tile 128(M) x 128(N), k-step 32.
//   A/B tiles double-buffered in LDS (f32), filled by the Tensor Data Mover
//   (fallback: cooperative b128 loads). Wave tile 64x32 = 4x2 WMMAs
//   (v_wmma_f32_16x16x32_bf16), f32->bf16 packed in registers at frag load.
// ---------------------------------------------------------------------------
__global__ __launch_bounds__(256)
void k_gemm_tanh(const float* __restrict__ X, const float* __restrict__ W,
                 const float* __restrict__ b_ih, const float* __restrict__ b_hh,
                 float* __restrict__ rnn_out) {
  __shared__ __align__(16) float lA[2][BM][BK];   // 2 x 16 KB
  __shared__ __align__(16) float lB[2][BN][BK];   // 2 x 16 KB

  const int lane  = threadIdx.x & 31;
  const int wv    = threadIdx.x >> 5;
  const int waveM = wv & 1;                        // 0..1 -> 64-row slab
  const int waveN = wv >> 1;                       // 0..3 -> 32-col slab
  const int blockM = blockIdx.y * BM;
  const int blockN = blockIdx.x * BN;
  const int m0 = waveM * 64;
  const int n0 = waveN * 32;
  const int r = lane & 15;                         // row/col within 16-tile
  const int g = lane >> 4;                         // lane-group (K split)

  v8f acc[4][2];
#pragma unroll
  for (int mi = 0; mi < 4; ++mi)
#pragma unroll
    for (int nj = 0; nj < 2; ++nj)
      acc[mi][nj] = (v8f){0.f, 0.f, 0.f, 0.f, 0.f, 0.f, 0.f, 0.f};

#if USE_TDM
  // Preload k-tile 0 into buffer 0 (one TDM per matrix tile, wave 0 issues).
  if (wv == 0) {
    tdm_load_tile_f32(X + (size_t)blockM * INPUT, (void*)&lA[0][0][0], BM, INPUT);
    tdm_load_tile_f32(W + (size_t)blockN * INPUT, (void*)&lB[0][0][0], BN, INPUT);
  }
#endif

  for (int kt = 0; kt < INPUT / BK; ++kt) {
    const int cur = kt & 1;

#if USE_TDM
    if (wv == 0) __builtin_amdgcn_s_wait_tensorcnt(0);   // cur buffer landed
    __syncthreads();                                     // publish to all waves
    if (wv == 0 && (kt + 1) < INPUT / BK) {              // prefetch next buffer
      const int kn = (kt + 1) * BK;
      tdm_load_tile_f32(X + (size_t)blockM * INPUT + kn,
                        (void*)&lA[cur ^ 1][0][0], BM, INPUT);
      tdm_load_tile_f32(W + (size_t)blockN * INPUT + kn,
                        (void*)&lB[cur ^ 1][0][0], BN, INPUT);
    }
#else
    const int k0 = kt * BK;
    __syncthreads();                                     // prior reads done
    {
      // Cooperative fill: 256 threads x 16 f32 per tile.
      const int row = threadIdx.x >> 1;
      const int c0  = (threadIdx.x & 1) * 16;
      const float* gx = X + (size_t)(blockM + row) * INPUT + k0 + c0;
      const float* gw = W + (size_t)(blockN + row) * INPUT + k0 + c0;
      float4* da = (float4*)&lA[cur][row][c0];
      float4* db = (float4*)&lB[cur][row][c0];
#pragma unroll
      for (int q = 0; q < 4; ++q) da[q] = ((const float4*)gx)[q];
#pragma unroll
      for (int q = 0; q < 4; ++q) db[q] = ((const float4*)gw)[q];
      if (k0 + BK < INPUT) {
        __builtin_prefetch(gx + BK, 0, 3);
        __builtin_prefetch(gw + BK, 0, 3);
      }
    }
    __syncthreads();
#endif

    // Fragment loads from LDS (f32 -> packed bf16):
    //   A: lane r holds row M=r; halfs 0..7 = K 8g..8g+7, 8..15 = K 16+8g..
    //   B: lane r holds col N=r; halfs 0..7 = K 8g..8g+7, 8..15 = K 16+8g..
    v16bf afr[4], bfr[2];
#pragma unroll
    for (int mi = 0; mi < 4; ++mi) {
      const float* pa = &lA[cur][m0 + mi * 16 + r][0];
      float4 q0 = *(const float4*)(pa + 8 * g);
      float4 q1 = *(const float4*)(pa + 8 * g + 4);
      float4 q2 = *(const float4*)(pa + 16 + 8 * g);
      float4 q3 = *(const float4*)(pa + 20 + 8 * g);
      cvt8(afr[mi], 0, q0, q1);
      cvt8(afr[mi], 8, q2, q3);
    }
#pragma unroll
    for (int nj = 0; nj < 2; ++nj) {
      const float* pb = &lB[cur][n0 + nj * 16 + r][0];
      float4 q0 = *(const float4*)(pb + 8 * g);
      float4 q1 = *(const float4*)(pb + 8 * g + 4);
      float4 q2 = *(const float4*)(pb + 16 + 8 * g);
      float4 q3 = *(const float4*)(pb + 20 + 8 * g);
      cvt8(bfr[nj], 0, q0, q1);
      cvt8(bfr[nj], 8, q2, q3);
    }

#pragma unroll
    for (int mi = 0; mi < 4; ++mi)
#pragma unroll
      for (int nj = 0; nj < 2; ++nj)
        acc[mi][nj] = __builtin_amdgcn_wmma_f32_16x16x32_bf16(
            /*neg_a=*/false, afr[mi], /*neg_b=*/false, bfr[nj],
            /*c_mod=*/(short)0, acc[mi][nj],
            /*reuse_a=*/false, /*reuse_b=*/false);

#if !USE_TDM
    // (barrier at the top of next iteration protects the single live buffer)
#endif
  }

  // Epilogue: bias + tanh. C layout: elem i -> row = 8*(lane>=16)+i, col = lane%16.
#pragma unroll
  for (int nj = 0; nj < 2; ++nj) {
    const int col  = blockN + n0 + nj * 16 + r;
    const float bias = b_ih[col] + b_hh[col];
#pragma unroll
    for (int mi = 0; mi < 4; ++mi) {
#pragma unroll
      for (int i = 0; i < 8; ++i) {
        const int row = blockM + m0 + mi * 16 + g * 8 + i;
        rnn_out[(size_t)row * HIDDEN + col] = tanhf(acc[mi][nj][i] + bias);
      }
    }
  }
}

// ---------------------------------------------------------------------------
// Kernel 2: AvgPool1d(kernel=45, stride=45) over hidden dim -> pooled (B, 45)
// ---------------------------------------------------------------------------
__global__ __launch_bounds__(256)
void k_pool(const float* __restrict__ rnn_out, float* __restrict__ pooled) {
  const int idx = blockIdx.x * blockDim.x + threadIdx.x;
  if (idx >= B_ROWS * NUM_EH) return;
  const int b = idx / NUM_EH;
  const int f = idx % NUM_EH;
  const float* p = rnn_out + (size_t)b * HIDDEN + f * KPOOL;
  float s = 0.f;
#pragma unroll
  for (int j = 0; j < KPOOL; ++j) s += p[j];
  pooled[idx] = s * (1.0f / (float)KPOOL);
}

// ---------------------------------------------------------------------------
// Kernel 3: inclusive prefix sums (cs, cs2) over the batch axis, one block
// per feature. LDS Hillis-Steele scan over 32 chunks of 256 with carry.
// ---------------------------------------------------------------------------
__global__ __launch_bounds__(256)
void k_scan(const float* __restrict__ pooled,
            float* __restrict__ cs, float* __restrict__ cs2) {
  __shared__ float s1[256];
  __shared__ float s2[256];
  const int f   = blockIdx.x;
  const int tid = threadIdx.x;
  float carry1 = 0.f, carry2 = 0.f;
  for (int chunk = 0; chunk < B_ROWS / 256; ++chunk) {
    const int b = chunk * 256 + tid;
    const float v  = pooled[(size_t)b * NUM_EH + f];
    s1[tid] = v;
    s2[tid] = v * v;
    __syncthreads();
#pragma unroll
    for (int off = 1; off < 256; off <<= 1) {
      const float t1 = (tid >= off) ? s1[tid - off] : 0.f;
      const float t2 = (tid >= off) ? s2[tid - off] : 0.f;
      __syncthreads();
      s1[tid] += t1;
      s2[tid] += t2;
      __syncthreads();
    }
    cs [(size_t)b * NUM_EH + f] = s1[tid] + carry1;
    cs2[(size_t)b * NUM_EH + f] = s2[tid] + carry2;
    const float tot1 = s1[255];
    const float tot2 = s2[255];
    __syncthreads();
    carry1 += tot1;
    carry2 += tot2;
  }
}

// ---------------------------------------------------------------------------
// Kernel 4: windowed mean/var from prefix sums -> est (B, 270)
//   est[b, (f*3 + j)*2 + {0,1}] = mean, var for window L_j ending at b.
// ---------------------------------------------------------------------------
__global__ __launch_bounds__(256)
void k_est(const float* __restrict__ cs, const float* __restrict__ cs2,
           float* __restrict__ est) {
  const int idx = blockIdx.x * blockDim.x + threadIdx.x;
  if (idx >= B_ROWS * NUM_EH * NJ) return;
  const int j = idx % NJ;
  const int f = (idx / NJ) % NUM_EH;
  const int b = idx / (NJ * NUM_EH);
  const int L = (j == 0) ? 32 : ((j == 1) ? 128 : 512);

  const size_t p  = (size_t)b * NUM_EH + f;
  const float c1  = cs[p];
  const float c2  = cs2[p];
  float c1l = 0.f, c2l = 0.f;
  if (b >= L) {
    const size_t pl = (size_t)(b - L) * NUM_EH + f;
    c1l = cs[pl];
    c2l = cs2[pl];
  }
  const int   n    = (b + 1 < L) ? (b + 1) : L;
  const float inv  = 1.0f / (float)n;
  const float m    = (c1 - c1l) * inv;
  const float var  = (c2 - c2l) * inv - m * m;

  const size_t o = (size_t)b * EST_DIM + ((size_t)f * NJ + j) * 2;
  est[o]     = m;
  est[o + 1] = var;
}

// ---------------------------------------------------------------------------
// Kernel 5: logits = [rnn_out | est] @ W_lin^T + b_lin; softmax over 10.
// One wave32 per batch row; 10 accumulators per lane; shfl_xor reduce.
// ---------------------------------------------------------------------------
__global__ __launch_bounds__(256)
void k_head(const float* __restrict__ rnn_out, const float* __restrict__ est,
            const float* __restrict__ W_lin, const float* __restrict__ b_lin,
            float* __restrict__ out) {
  const int lane = threadIdx.x & 31;
  const int wave = threadIdx.x >> 5;
  const int b = blockIdx.x * 8 + wave;
  if (b >= B_ROWS) return;

  const float* rrow = rnn_out + (size_t)b * HIDDEN;
  const float* erow = est     + (size_t)b * EST_DIM;

  float acc[CLASSES];
#pragma unroll
  for (int c = 0; c < CLASSES; ++c) acc[c] = 0.f;

  for (int k = lane; k < LIN_DIM; k += 32) {
    const float x = (k < HIDDEN) ? rrow[k] : erow[k - HIDDEN];
#pragma unroll
    for (int c = 0; c < CLASSES; ++c)
      acc[c] += x * W_lin[(size_t)c * LIN_DIM + k];
  }
#pragma unroll
  for (int c = 0; c < CLASSES; ++c) acc[c] = wave_reduce_add(acc[c]);

  if (lane == 0) {
    float logit[CLASSES];
    float mx = -INFINITY;
#pragma unroll
    for (int c = 0; c < CLASSES; ++c) {
      logit[c] = acc[c] + b_lin[c];
      mx = fmaxf(mx, logit[c]);
    }
    float s = 0.f;
#pragma unroll
    for (int c = 0; c < CLASSES; ++c) {
      logit[c] = __expf(logit[c] - mx);
      s += logit[c];
    }
    const float inv = 1.0f / s;
#pragma unroll
    for (int c = 0; c < CLASSES; ++c)
      out[(size_t)b * CLASSES + c] = logit[c] * inv;
  }
}

// ---------------------------------------------------------------------------
// Launch
// ---------------------------------------------------------------------------
extern "C" void kernel_launch(void* const* d_in, const int* in_sizes, int n_in,
                              void* d_out, int out_size, void* d_ws, size_t ws_size,
                              hipStream_t stream) {
  const float* x     = (const float*)d_in[0];   // (B, 1, 2048)
  const float* W_ih  = (const float*)d_in[1];   // (2048, 2048)
  // d_in[2] = W_hh : unused (seq_len==1, h0==0 -> recurrent term is zero)
  const float* b_ih  = (const float*)d_in[3];
  const float* b_hh  = (const float*)d_in[4];
  const float* W_lin = (const float*)d_in[5];   // (10, 2318)
  const float* b_lin = (const float*)d_in[6];
  float* out = (float*)d_out;

  float* ws      = (float*)d_ws;
  float* rnn_out = ws;                                          // B*2048
  float* pooled  = rnn_out + (size_t)B_ROWS * HIDDEN;           // B*45
  float* cs      = pooled  + (size_t)B_ROWS * NUM_EH;           // B*45
  float* cs2     = cs      + (size_t)B_ROWS * NUM_EH;           // B*45
  float* est     = cs2     + (size_t)B_ROWS * NUM_EH;           // B*270

  dim3 g1(HIDDEN / BN, B_ROWS / BM);
  k_gemm_tanh<<<g1, 256, 0, stream>>>(x, W_ih, b_ih, b_hh, rnn_out);

  k_pool<<<(B_ROWS * NUM_EH + 255) / 256, 256, 0, stream>>>(rnn_out, pooled);

  k_scan<<<NUM_EH, 256, 0, stream>>>(pooled, cs, cs2);

  k_est<<<(B_ROWS * NUM_EH * NJ + 255) / 256, 256, 0, stream>>>(cs, cs2, est);

  k_head<<<B_ROWS / 8, 256, 0, stream>>>(rnn_out, est, W_lin, b_lin, out);
}
